// SPVNAS_CNN_79602923864080
// MI455X (gfx1250) — compile-verified, matrix operand order checked by reading, not run on hardware
//
#include <hip/hip_runtime.h>
#include <hip/hip_bf16.h>
#include <stdint.h>

// ---------------------------------------------------------------------------
// SPVNAS sparse U-Net forward for MI455X (gfx1250, wave32, WMMA).
// All GEMM-like stages (rulebook sparse conv, transposed conv, 1x1 shortcut,
// dense point MLPs, classifier) use ONE generic WMMA kernel built on
// v_wmma_f32_16x16x32_f16.  Operands are loaded DIRECTLY into fragment
// registers (no LDS): features live in padded f16 shadow buffers, weights are
// pre-swizzled on device into exact CDNA5 fragment order.  Rulebook masks are
// applied per-lane (each lane owns one A-matrix row in the 16-bit layout).
// ---------------------------------------------------------------------------

typedef _Float16 h16;
typedef __attribute__((ext_vector_type(8)))  _Float16 v8h;
typedef __attribute__((ext_vector_type(16))) _Float16 v16h;
typedef __attribute__((ext_vector_type(8)))  float    v8f;

#define P_   400000
#define N0_  300000
#define N1_  100000
#define N2_  30000
#define NCLS_ 19

enum { MODE_NBR = 0, MODE_TCONV = 1, MODE_IDENT = 2 };
enum { ACT_NONE = 0, ACT_RELU = 1, ACT_LEAKY = 2 };

// ---------------------------------------------------------------------------
// Generic fused sparse-conv / GEMM kernel.  One wave = one 16x16 output tile.
//   acc = sum_k sum_cc  A(gather(nbr,k), f16, masked) x B(W[k], f16)
//   out = act(acc*scale + shift + resid)   -> fp32 Y and/or f16 shadow Y16
// A-fragment: lane owns row M=lane&15, K-slices {kb..kb+7, kb+16..kb+23},
// kb = 8*(lane>=16)  ->  two contiguous 16B loads from the f16 feature row.
// B-fragment: pre-swizzled weight blocks, 32B contiguous per lane.
// ---------------------------------------------------------------------------
__global__ __launch_bounds__(128) void spconv_wmma_kernel(
    const h16* __restrict__ X16, int cinp,          // padded f16 features
    const h16* __restrict__ Wsw, int nC0, int nCT,  // swizzled weights
    int K, int mode,
    const int*   __restrict__ nbr,   // MODE_NBR: [n_out,K]  MODE_TCONV: parent
    const float* __restrict__ mask,  // MODE_NBR: [n_out,K]
    const int*   __restrict__ off,   // MODE_TCONV: [n_out] in [0,K)
    const float* __restrict__ scale, const float* __restrict__ shift,
    const float* __restrict__ resid, int resid_ld,
    float* __restrict__ Y,  int ldY,  int col_off,
    h16*   __restrict__ Y16, int ldY16, int col_off16,
    int cout, int act, int n_out)
{
    const int wave = threadIdx.x >> 5;
    const int lane = threadIdx.x & 31;
    const int ct = blockIdx.y * 4 + wave;
    if (ct >= nCT) return;                 // wave-uniform exit

    const int rowTile = blockIdx.x << 4;
    const int mr = lane & 15;              // matrix row owned by this lane
    const int kb = (lane >> 4) << 3;       // K-slice base: 0 or 8
    const int rowA  = rowTile + mr;
    const int valid = (rowA < n_out);
    const int rowc  = valid ? rowA : 0;

    v8f acc = {};

    for (int k = 0; k < K; ++k) {
        int g; float m;
        if (mode == MODE_NBR) {
            g = nbr[(size_t)rowc * K + k];
            m = mask[(size_t)rowc * K + k];
        } else if (mode == MODE_TCONV) {
            g = nbr[rowc];
            m = (off[rowc] == k) ? 1.f : 0.f;
        } else {
            g = rowc; m = 1.f;
        }
        if (!valid) m = 0.f;
        const h16  mh = (h16)m;
        const h16* __restrict__ xr = X16 + (size_t)g * cinp;
        const h16* __restrict__ wb = Wsw + (((size_t)k * nCT + ct) * nC0) * 512
                                         + (lane << 4);
        for (int cc = 0; cc < nC0; ++cc) {
            const v8h alo = *(const v8h*)(xr + (cc << 5) + kb);
            const v8h ahi = *(const v8h*)(xr + (cc << 5) + kb + 16);
            v16h a = __builtin_shufflevector(alo, ahi,
                0,1,2,3,4,5,6,7,8,9,10,11,12,13,14,15);
            #pragma unroll
            for (int j = 0; j < 16; ++j) a[j] *= mh;   // per-lane rulebook mask
            const v8h blo = *(const v8h*)(wb + (size_t)cc * 512);
            const v8h bhi = *(const v8h*)(wb + (size_t)cc * 512 + 8);
            const v16h b = __builtin_shufflevector(blo, bhi,
                0,1,2,3,4,5,6,7,8,9,10,11,12,13,14,15);
            acc = __builtin_amdgcn_wmma_f32_16x16x32_f16(
                false, a, false, b, (short)0, acc, false, false);
        }
    }

    // ---- epilogue: BN + bias + residual + activation (documented C/D layout)
    const int sel  = lane >> 4;
    const int ncol = lane & 15;
    const int colG = (ct << 4) + ncol;
    if (colG < cout) {
        const float s  = scale ? scale[colG] : 1.f;
        const float t0 = shift ? shift[colG] : 0.f;
        #pragma unroll
        for (int i = 0; i < 8; ++i) {
            int row = rowTile + (sel << 3) + i;
            if (row < n_out) {
                float v = acc[i] * s + t0;
                if (resid) v += resid[(size_t)row * resid_ld + colG];
                if (act == ACT_RELU)       v = fmaxf(v, 0.f);
                else if (act == ACT_LEAKY) v = (v > 0.f) ? v : 0.1f * v;
                if (Y)   Y[(size_t)row * ldY + col_off + colG] = v;
                if (Y16) Y16[(size_t)row * ldY16 + col_off16 + colG] = (h16)v;
            }
        }
    }
}

// ---------------------------------------------------------------------------
// Support kernels
// ---------------------------------------------------------------------------
__global__ void zero_kernel(float* p, size_t n) {
    size_t t = (size_t)blockIdx.x * blockDim.x + threadIdx.x;
    if (t < n) p[t] = 0.f;
}
__global__ void zero16_kernel(h16* p, size_t n) {
    size_t t = (size_t)blockIdx.x * blockDim.x + threadIdx.x;
    if (t < n) p[t] = (h16)0.f;
}

__global__ void scatter_add_kernel(const float* __restrict__ X, const int* __restrict__ idx,
                                   float* acc, float* cnt, int Pn, int C) {
    size_t t = (size_t)blockIdx.x * blockDim.x + threadIdx.x;
    if (t >= (size_t)Pn * C) return;
    int p = (int)(t / C), c = (int)(t % C);
    int row = idx[p];
    atomicAdd(&acc[(size_t)row * C + c], X[t]);
    if (c == 0) atomicAdd(&cnt[row], 1.f);
}

// divide by counts; emit padded f16 shadow
__global__ void seg_div_kernel(float* acc, const float* __restrict__ cnt, int n, int C,
                               h16* __restrict__ out16, int ld16) {
    size_t t = (size_t)blockIdx.x * blockDim.x + threadIdx.x;
    if (t >= (size_t)n * C) return;
    int r = (int)(t / C), c = (int)(t % C);
    float v = acc[t] / fmaxf(cnt[r], 1.f);
    acc[t] = v;
    if (out16) out16[(size_t)r * ld16 + c] = (h16)v;
}

// trilinear voxel->point: out[p,c] (+)= sum_j w[p,j]*XV[idx[p,j],c]; f32 + f16
__global__ void devox_kernel(const float* __restrict__ XV, const int* __restrict__ idx,
                             const float* __restrict__ w, float* __restrict__ out,
                             h16* __restrict__ out16, int ld16,
                             int Pn, int C, int accumulate) {
    size_t t = (size_t)blockIdx.x * blockDim.x + threadIdx.x;
    if (t >= (size_t)Pn * C) return;
    int p = (int)(t / C), c = (int)(t % C);
    const int*   ip = idx + (size_t)p * 8;
    const float* wp = w   + (size_t)p * 8;
    float s = 0.f;
    #pragma unroll
    for (int j = 0; j < 8; ++j) s += wp[j] * XV[(size_t)ip[j] * C + c];
    if (accumulate) s += out[t];
    out[t] = s;
    if (out16) out16[(size_t)p * ld16 + c] = (h16)s;
}

// skip-concat: copy fp32 columns into f32 and/or f16 destination windows
__global__ void copy_cols_kernel(const float* __restrict__ X, int cin,
                                 float* __restrict__ Y, int ldY, int coloff,
                                 h16* __restrict__ Y16, int ld16, int coloff16,
                                 int rows) {
    size_t t = (size_t)blockIdx.x * blockDim.x + threadIdx.x;
    if (t >= (size_t)rows * cin) return;
    int r = (int)(t / cin), c = (int)(t % cin);
    float v = X[t];
    if (Y)   Y[(size_t)r * ldY + coloff + c] = v;
    if (Y16) Y16[(size_t)r * ld16 + coloff16 + c] = (h16)v;
}

// fp32 [K,cin,cout] -> f16 swizzled fragment blocks:
//   block(k, ct, cc) = 512 halfs, lane-major (32 lanes x 16 halfs, 32B/lane)
//   lane: mr=lane&15 (N), kb=8*(lane>=16); j<8 -> K=cc*32+kb+j ; j>=8 -> +16
__global__ void cvt_w_swz_kernel(const float* __restrict__ W, h16* __restrict__ Wsw,
                                 int K, int cin, int cout, int nC0, int nCT) {
    size_t n = (size_t)K * nCT * nC0 * 512;
    size_t t = (size_t)blockIdx.x * blockDim.x + threadIdx.x;
    if (t >= n) return;
    int j    = (int)(t & 15);
    int lane = (int)((t >> 4) & 31);
    size_t blk = t >> 9;
    int cc = (int)(blk % nC0);
    size_t r2 = blk / nC0;
    int ct = (int)(r2 % nCT);
    int k  = (int)(r2 / nCT);
    int mr = lane & 15, kb = (lane >> 4) << 3;
    int kk = (j < 8) ? (kb + j) : (kb + 16 + (j - 8));
    int c  = (cc << 5) + kk;
    int nn = (ct << 4) + mr;
    Wsw[t] = (c < cin && nn < cout) ? (h16)W[((size_t)k * cin + c) * cout + nn]
                                    : (h16)0.f;
}

// Fold BN(+bias) into scale/shift:  y = x*s + sh
__global__ void prep_affine_kernel(const float* g, const float* b, const float* m,
                                   const float* v, const float* bias,
                                   float* scale, float* shift, int n) {
    int t = blockIdx.x * blockDim.x + threadIdx.x;
    if (t >= n) return;
    float s  = g ? g[t] * rsqrtf(v[t] + 1e-5f) : 1.f;
    float sh = b ? (b[t] - m[t] * s) : 0.f;
    if (bias) sh += bias[t] * s;
    scale[t] = s; shift[t] = sh;
}

// ---------------------------------------------------------------------------
extern "C" void kernel_launch(void* const* d_in, const int* in_sizes, int n_in,
                              void* d_out, int out_size, void* d_ws, size_t ws_size,
                              hipStream_t stream) {
    (void)in_sizes; (void)n_in; (void)out_size; (void)ws_size;

    // -------- unpack inputs (setup_inputs dict insertion order, recursive) ----
    int ii = 0;
    auto nf = [&]() { return (const float*)d_in[ii++]; };
    auto ni = [&]() { return (const int*)d_in[ii++]; };
    struct BN { const float *g, *b, *m, *v; };
    auto nbn = [&]() { BN q; q.g = nf(); q.b = nf(); q.m = nf(); q.v = nf(); return q; };

    const float* pts = nf();
    const float* stem1_W = nf(); BN stem1_bn = nbn();
    const float* stem2_W = nf(); BN stem2_bn = nbn();
    const float* s1dn_W  = nf(); BN s1dn_bn  = nbn();
    const float* s1r_W1  = nf(); BN s1r_bn1  = nbn();
    const float* s1r_W2  = nf(); BN s1r_bn2  = nbn();
    const float* s1r_Wd  = nf(); BN s1r_bnd  = nbn();
    const float* s4dn_W  = nf(); BN s4dn_bn  = nbn();
    const float* s4r_W1  = nf(); BN s4r_bn1  = nbn();
    const float* s4r_W2  = nf(); BN s4r_bn2  = nbn();
    const float* s4r_Wd  = nf(); BN s4r_bnd  = nbn();
    const float* u3_W    = nf(); BN u3_bn    = nbn();
    const float* u3r_W1  = nf(); BN u3r_bn1  = nbn();
    const float* u3r_W2  = nf(); BN u3r_bn2  = nbn();
    const float* u4_W    = nf(); BN u4_bn    = nbn();
    const float* u4r_W1  = nf(); BN u4r_bn1  = nbn();
    const float* u4r_W2  = nf(); BN u4r_bn2  = nbn();
    const float* u4r_Wd  = nf(); BN u4r_bnd  = nbn();
    const float* pt0_W = nf(); const float* pt0_b = nf(); BN pt0_bn = nbn();
    const float* pt2_W = nf(); const float* pt2_b = nf(); BN pt2_bn = nbn();
    const float* cls_W = nf(); const float* cls_b = nf();
    const int* p2v0 = ni(); const int* p2v2 = ni();
    const int* nbr0 = ni(); const float* mask0 = nf();
    const int* nbr1 = ni(); const float* mask1 = nf();
    const int* nbr2 = ni(); const float* mask2 = nf();
    const int* d1_nbr = ni(); const float* d1_mask = nf();
    const int* d2_nbr = ni(); const float* d2_mask = nf();
    const int* v2p0_idx = ni(); const float* v2p0_w = nf();
    const int* v2p2_idx = ni(); const float* v2p2_w = nf();
    const int* u1_par = ni(); const int* u1_off = ni();
    const int* u0_par = ni(); const int* u0_off = ni();

    // -------- workspace bump allocator ---------------------------------------
    uint8_t* wsb = (uint8_t*)d_ws;
    size_t wo = 0;
    auto alloc  = [&](size_t bytes) { void* p = wsb + wo; wo = (wo + bytes + 255) & ~(size_t)255; return p; };
    auto allocF = [&](size_t n) { return (float*)alloc(n * sizeof(float)); };
    auto blocks = [](size_t n) { return dim3((unsigned)((n + 255) / 256)); };
    auto allocH = [&](size_t n, bool clear) {
        h16* p = (h16*)alloc(n * sizeof(h16));
        if (clear) zero16_kernel<<<blocks(n), 256, 0, stream>>>(p, n);
        return p;
    };

    // -------- swizzled f16 weights + BN folding ------------------------------
    auto cvtW = [&](const float* W, int K, int cin, int cout, int cinp, int coutp) {
        int nC0 = cinp / 32, nCT = coutp / 16;
        size_t n = (size_t)K * nCT * nC0 * 512;
        h16* Wsw = (h16*)alloc(n * sizeof(h16));
        cvt_w_swz_kernel<<<blocks(n), 256, 0, stream>>>(W, Wsw, K, cin, cout, nC0, nCT);
        return Wsw;
    };
    auto aff = [&](const float* g, const float* b, const float* m, const float* v,
                   const float* bias, int n) {
        float* A = allocF((size_t)2 * n);
        prep_affine_kernel<<<dim3((unsigned)((n + 63) / 64)), 64, 0, stream>>>(
            g, b, m, v, bias, A, A + n, n);
        return A;
    };

    h16* Wstem1 = cvtW(stem1_W, 27,   5,  32,  32,  32);
    h16* Wstem2 = cvtW(stem2_W, 27,  32,  32,  32,  32);
    h16* Ws1dn  = cvtW(s1dn_W,   8,  32,  16,  32,  16);
    h16* Ws1r1  = cvtW(s1r_W1,  27,  16,  20,  32,  32);
    h16* Ws1r2  = cvtW(s1r_W2,  27,  20,  28,  32,  32);
    h16* Ws1rd  = cvtW(s1r_Wd,   1,  16,  28,  32,  32);
    h16* Ws4dn  = cvtW(s4dn_W,   8,  28, 108,  32, 112);
    h16* Ws4r1  = cvtW(s4r_W1,  27, 108, 108, 128, 112);
    h16* Ws4r2  = cvtW(s4r_W2,  27, 108, 156, 128, 160);
    h16* Ws4rd  = cvtW(s4r_Wd,   1, 108, 156, 128, 160);
    h16* Wu3    = cvtW(u3_W,     8, 156,  32, 160,  32);
    h16* Wu3r1  = cvtW(u3r_W1,  27,  60,  40,  64,  48);
    h16* Wu3r2  = cvtW(u3r_W2,  27,  40,  60,  64,  64);
    h16* Wu4    = cvtW(u4_W,     8,  60,  44,  64,  48);
    h16* Wu4r1  = cvtW(u4r_W1,  27,  76,  40,  96,  48);
    h16* Wu4r2  = cvtW(u4r_W2,  27,  40,  48,  64,  48);
    h16* Wu4rd  = cvtW(u4r_Wd,   1,  76,  48,  96,  48);
    h16* Wpt0   = cvtW(pt0_W,    1,  32, 156,  32, 160);
    h16* Wpt2   = cvtW(pt2_W,    1, 156,  48, 160,  48);
    h16* Wcls   = cvtW(cls_W,    1,  48, NCLS_, 64, 32);

    float* Astem1 = aff(stem1_bn.g, stem1_bn.b, stem1_bn.m, stem1_bn.v, nullptr, 32);
    float* Astem2 = aff(stem2_bn.g, stem2_bn.b, stem2_bn.m, stem2_bn.v, nullptr, 32);
    float* As1dn  = aff(s1dn_bn.g,  s1dn_bn.b,  s1dn_bn.m,  s1dn_bn.v,  nullptr, 16);
    float* As1r1  = aff(s1r_bn1.g, s1r_bn1.b, s1r_bn1.m, s1r_bn1.v, nullptr, 20);
    float* As1r2  = aff(s1r_bn2.g, s1r_bn2.b, s1r_bn2.m, s1r_bn2.v, nullptr, 28);
    float* As1rd  = aff(s1r_bnd.g, s1r_bnd.b, s1r_bnd.m, s1r_bnd.v, nullptr, 28);
    float* As4dn  = aff(s4dn_bn.g, s4dn_bn.b, s4dn_bn.m, s4dn_bn.v, nullptr, 108);
    float* As4r1  = aff(s4r_bn1.g, s4r_bn1.b, s4r_bn1.m, s4r_bn1.v, nullptr, 108);
    float* As4r2  = aff(s4r_bn2.g, s4r_bn2.b, s4r_bn2.m, s4r_bn2.v, nullptr, 156);
    float* As4rd  = aff(s4r_bnd.g, s4r_bnd.b, s4r_bnd.m, s4r_bnd.v, nullptr, 156);
    float* Au3    = aff(u3_bn.g, u3_bn.b, u3_bn.m, u3_bn.v, nullptr, 32);
    float* Au3r1  = aff(u3r_bn1.g, u3r_bn1.b, u3r_bn1.m, u3r_bn1.v, nullptr, 40);
    float* Au3r2  = aff(u3r_bn2.g, u3r_bn2.b, u3r_bn2.m, u3r_bn2.v, nullptr, 60);
    float* Au4    = aff(u4_bn.g, u4_bn.b, u4_bn.m, u4_bn.v, nullptr, 44);
    float* Au4r1  = aff(u4r_bn1.g, u4r_bn1.b, u4r_bn1.m, u4r_bn1.v, nullptr, 40);
    float* Au4r2  = aff(u4r_bn2.g, u4r_bn2.b, u4r_bn2.m, u4r_bn2.v, nullptr, 48);
    float* Au4rd  = aff(u4r_bnd.g, u4r_bnd.b, u4r_bnd.m, u4r_bnd.v, nullptr, 48);
    float* Apt0   = aff(pt0_bn.g, pt0_bn.b, pt0_bn.m, pt0_bn.v, pt0_b, 156);
    float* Apt2   = aff(pt2_bn.g, pt2_bn.b, pt2_bn.m, pt2_bn.v, pt2_b, 48);
    float* Acls   = aff(nullptr, nullptr, nullptr, nullptr, cls_b, NCLS_);

    // -------- fp32 buffers (only where fp32 consumers exist) -----------------
    float* cnt0 = allocF(N0_);
    float* cnt2 = allocF(N2_);
    float* v5   = allocF((size_t)N0_ * 5);     // segmean scratch
    float* xv0  = allocF((size_t)N0_ * 32);    // segmean scratch
    float* y3v  = allocF((size_t)N2_ * 156);   // segmean scratch
    float* x0   = allocF((size_t)N0_ * 32);    // devox src + concat src
    float* z0   = allocF((size_t)P_  * 32);    // segmean src
    float* sc1  = allocF((size_t)N1_ * 28);
    float* x1   = allocF((size_t)N1_ * 28);    // concat src
    float* sc4  = allocF((size_t)N2_ * 156);
    float* x4   = allocF((size_t)N2_ * 156);   // devox src
    float* z1   = allocF((size_t)P_  * 156);   // accum + segmean src
    float* y3t  = allocF((size_t)N1_ * 60);    // resid
    float* sc4b = allocF((size_t)N0_ * 48);
    float* y4   = allocF((size_t)N0_ * 48);    // devox src
    float* z3   = allocF((size_t)P_  * 48);    // accum src

    // -------- f16 shadow buffers (padded; pre-zeroed where padding exists) ---
    h16* v5h   = allocH((size_t)N0_ * 32,  true);   // 5 -> 32
    h16* x0ah  = allocH((size_t)N0_ * 32,  false);  // full 32
    h16* xv0h  = allocH((size_t)N0_ * 32,  false);
    h16* z0h   = allocH((size_t)P_  * 32,  false);
    h16* x1dh  = allocH((size_t)N1_ * 32,  true);   // 16 -> 32
    h16* h1ah  = allocH((size_t)N1_ * 32,  true);   // 20 -> 32
    h16* x1h   = allocH((size_t)N1_ * 32,  true);   // 28 -> 32
    h16* x4dh  = allocH((size_t)N2_ * 128, true);   // 108 -> 128
    h16* h4ah  = allocH((size_t)N2_ * 128, true);
    h16* z1h   = allocH((size_t)P_  * 160, true);   // 156 -> 160
    h16* y3vh  = allocH((size_t)N2_ * 160, true);
    h16* y3th  = allocH((size_t)N1_ * 64,  true);   // 60 -> 64
    h16* h3ah  = allocH((size_t)N1_ * 64,  true);   // 40 -> 64
    h16* y3h   = allocH((size_t)N1_ * 64,  true);   // 60 -> 64
    h16* y4th  = allocH((size_t)N0_ * 96,  true);   // 76 -> 96
    h16* h4bh  = allocH((size_t)N0_ * 64,  true);   // 40 -> 64
    h16* z3h   = allocH((size_t)P_  * 64,  true);   // 48 -> 64

    // -------- launch helpers -------------------------------------------------
    auto conv = [&](const h16* X16, int cinp, const h16* Wsw, int coutp,
                    int K, int mode, const int* nbrp, const float* maskp, const int* offp,
                    const float* A, int cout, const float* resid, int resid_ld,
                    float* Y, int ldY, int coloff,
                    h16* Y16, int ldY16, int coloff16,
                    int act, int n_out) {
        int nC0 = cinp / 32, nCT = coutp / 16;
        dim3 grid((unsigned)((n_out + 15) / 16), (unsigned)((nCT + 3) / 4));
        spconv_wmma_kernel<<<grid, 128, 0, stream>>>(
            X16, cinp, Wsw, nC0, nCT, K, mode, nbrp, maskp, offp,
            A ? A : nullptr, A ? A + cout : nullptr, resid, resid_ld,
            Y, ldY, coloff, Y16, ldY16, coloff16, cout, act, n_out);
    };
    auto segmean = [&](const float* F, int Pn, int C, const int* idx, int n,
                       float* acc, float* cnt, h16* out16, int ld16) {
        zero_kernel<<<blocks((size_t)n * C), 256, 0, stream>>>(acc, (size_t)n * C);
        zero_kernel<<<blocks((size_t)n), 256, 0, stream>>>(cnt, (size_t)n);
        scatter_add_kernel<<<blocks((size_t)Pn * C), 256, 0, stream>>>(F, idx, acc, cnt, Pn, C);
        seg_div_kernel<<<blocks((size_t)n * C), 256, 0, stream>>>(acc, cnt, n, C, out16, ld16);
    };
    auto devox = [&](const float* XV, const int* idx, const float* w, float* out,
                     h16* out16, int ld16, int C, int accum) {
        devox_kernel<<<blocks((size_t)P_ * C), 256, 0, stream>>>(
            XV, idx, w, out, out16, ld16, P_, C, accum);
    };

    // ======================= forward pipeline ===============================
    // initial_voxelize + stem
    segmean(pts, P_, 5, p2v0, N0_, v5, cnt0, v5h, 32);
    conv(v5h, 32, Wstem1, 32, 27, MODE_NBR, nbr0, mask0, nullptr, Astem1, 32,
         nullptr, 0, nullptr, 0, 0, x0ah, 32, 0, ACT_RELU, N0_);
    conv(x0ah, 32, Wstem2, 32, 27, MODE_NBR, nbr0, mask0, nullptr, Astem2, 32,
         nullptr, 0, x0, 32, 0, nullptr, 0, 0, ACT_RELU, N0_);
    devox(x0, v2p0_idx, v2p0_w, z0, z0h, 32, 32, 0);       // z0 [P,32]
    segmean(z0, P_, 32, p2v0, N0_, xv0, cnt0, xv0h, 32);   // point_to_voxel

    // stage1
    conv(xv0h, 32, Ws1dn, 16, 8, MODE_NBR, d1_nbr, d1_mask, nullptr, As1dn, 16,
         nullptr, 0, nullptr, 0, 0, x1dh, 32, 0, ACT_RELU, N1_);
    conv(x1dh, 32, Ws1rd, 32, 1, MODE_IDENT, nullptr, nullptr, nullptr, As1rd, 28,
         nullptr, 0, sc1, 28, 0, nullptr, 0, 0, ACT_NONE, N1_);
    conv(x1dh, 32, Ws1r1, 32, 27, MODE_NBR, nbr1, mask1, nullptr, As1r1, 20,
         nullptr, 0, nullptr, 0, 0, h1ah, 32, 0, ACT_RELU, N1_);
    conv(h1ah, 32, Ws1r2, 32, 27, MODE_NBR, nbr1, mask1, nullptr, As1r2, 28,
         sc1, 28, x1, 28, 0, x1h, 32, 0, ACT_RELU, N1_);   // x1 [N1,28]

    // stage4
    conv(x1h, 32, Ws4dn, 112, 8, MODE_NBR, d2_nbr, d2_mask, nullptr, As4dn, 108,
         nullptr, 0, nullptr, 0, 0, x4dh, 128, 0, ACT_RELU, N2_);
    conv(x4dh, 128, Ws4rd, 160, 1, MODE_IDENT, nullptr, nullptr, nullptr, As4rd, 156,
         nullptr, 0, sc4, 156, 0, nullptr, 0, 0, ACT_NONE, N2_);
    conv(x4dh, 128, Ws4r1, 112, 27, MODE_NBR, nbr2, mask2, nullptr, As4r1, 108,
         nullptr, 0, nullptr, 0, 0, h4ah, 128, 0, ACT_RELU, N2_);
    conv(h4ah, 128, Ws4r2, 160, 27, MODE_NBR, nbr2, mask2, nullptr, As4r2, 156,
         sc4, 156, x4, 156, 0, nullptr, 0, 0, ACT_RELU, N2_);  // x4 [N2,156]

    // z1 = devox(x4) + relu(bn(pt0(z0)))
    conv(z0h, 32, Wpt0, 160, 1, MODE_IDENT, nullptr, nullptr, nullptr, Apt0, 156,
         nullptr, 0, z1, 156, 0, nullptr, 0, 0, ACT_RELU, P_);
    devox(x4, v2p2_idx, v2p2_w, z1, z1h, 160, 156, 1);

    // up3
    segmean(z1, P_, 156, p2v2, N2_, y3v, cnt2, y3vh, 160);
    conv(y3vh, 160, Wu3, 32, 8, MODE_TCONV, u1_par, nullptr, u1_off, Au3, 32,
         nullptr, 0, y3t, 60, 0, y3th, 64, 0, ACT_RELU, N1_);
    copy_cols_kernel<<<blocks((size_t)N1_ * 28), 256, 0, stream>>>(
        x1, 28, y3t, 60, 32, y3th, 64, 32, N1_);
    conv(y3th, 64, Wu3r1, 48, 27, MODE_NBR, nbr1, mask1, nullptr, Au3r1, 40,
         nullptr, 0, nullptr, 0, 0, h3ah, 64, 0, ACT_RELU, N1_);
    conv(h3ah, 64, Wu3r2, 64, 27, MODE_NBR, nbr1, mask1, nullptr, Au3r2, 60,
         y3t, 60, nullptr, 0, 0, y3h, 64, 0, ACT_RELU, N1_);   // y3 [N1,60]

    // up4
    conv(y3h, 64, Wu4, 48, 8, MODE_TCONV, u0_par, nullptr, u0_off, Au4, 44,
         nullptr, 0, nullptr, 0, 0, y4th, 96, 0, ACT_RELU, N0_);
    copy_cols_kernel<<<blocks((size_t)N0_ * 32), 256, 0, stream>>>(
        x0, 32, nullptr, 0, 0, y4th, 96, 44, N0_);
    conv(y4th, 96, Wu4rd, 48, 1, MODE_IDENT, nullptr, nullptr, nullptr, Au4rd, 48,
         nullptr, 0, sc4b, 48, 0, nullptr, 0, 0, ACT_NONE, N0_);
    conv(y4th, 96, Wu4r1, 48, 27, MODE_NBR, nbr0, mask0, nullptr, Au4r1, 40,
         nullptr, 0, nullptr, 0, 0, h4bh, 64, 0, ACT_RELU, N0_);
    conv(h4bh, 64, Wu4r2, 48, 27, MODE_NBR, nbr0, mask0, nullptr, Au4r2, 48,
         sc4b, 48, y4, 48, 0, nullptr, 0, 0, ACT_RELU, N0_);   // y4 [N0,48]

    // z3 = devox(y4) + relu(bn(pt2(z1))), classifier
    conv(z1h, 160, Wpt2, 48, 1, MODE_IDENT, nullptr, nullptr, nullptr, Apt2, 48,
         nullptr, 0, z3, 48, 0, nullptr, 0, 0, ACT_RELU, P_);
    devox(y4, v2p0_idx, v2p0_w, z3, z3h, 64, 48, 1);
    conv(z3h, 64, Wcls, 32, 1, MODE_IDENT, nullptr, nullptr, nullptr, Acls, NCLS_,
         nullptr, 0, (float*)d_out, NCLS_, 0, nullptr, 0, 0, ACT_LEAKY, P_);
}
